// stGCL_noimage_29850022707203
// MI455X (gfx1250) — compile-verified, matrix operand order checked by reading, not run on hardware
//
#include <hip/hip_runtime.h>
#include <hip/hip_bf16.h>
#include <math.h>

// ---------------------------------------------------------------------------
// stGCL GAT autoencoder forward on gfx1250 (MI455X).
// Dense GEMMs: v_wmma_f32_16x16x32_bf16, one 16x(16*NT) strip per wave32 so
// the A fragment is loaded once and reused across NT WMMAs (4x less A traffic).
// Sparse GAT softmax-aggregation: atomic max/add streaming.
// ---------------------------------------------------------------------------

typedef __attribute__((ext_vector_type(16))) __bf16 v16bf;
typedef __attribute__((ext_vector_type(8)))  __bf16 v8bf;
typedef __attribute__((ext_vector_type(8)))  float  v8f;

#define LEAKY_SLOPE 0.2f

// order-preserving float <-> uint encoding for atomicMax on f32
__device__ __forceinline__ unsigned fenc(float f) {
    unsigned u = __float_as_uint(f);
    return (u & 0x80000000u) ? ~u : (u | 0x80000000u);
}
__device__ __forceinline__ float fdec(unsigned u) {
    return __uint_as_float((u & 0x80000000u) ? (u & 0x7fffffffu) : ~u);
}

// ---------------------------------------------------------------------------
// C[M,NC] = A[M,K](f32, optionally row-permuted) * B[K,NC]
// B supplied as BT[NC,K] row-major bf16 (i.e. B column-major).
// Optional duplicate output C1 (for the doubled h4 slots).
// Each wave computes a 16 x (16*NT) strip: A fragment built once per K-step,
// NT back-to-back v_wmma_f32_16x16x32_bf16 against NT B fragments.
// ---------------------------------------------------------------------------
template <int NT>
__global__ void wmma_gemm_bf16(const float* __restrict__ A,
                               const __bf16* __restrict__ BT,
                               float* __restrict__ C0,
                               float* __restrict__ C1,
                               int M, int K, int NC, int usePerm)
{
    const int lane    = threadIdx.x & 31;
    const int wave    = blockIdx.x * (blockDim.x >> 5) + (threadIdx.x >> 5);
    const int groupsN = (NC >> 4) / NT;
    const int total   = (M >> 4) * groupsN;
    if (wave >= total) return;                 // wave-uniform: EXEC stays all-1s

    const int tileM = wave / groupsN;
    const int gN    = wave - tileM * groupsN;

    const int half = lane >> 4;                // K sub-chunk selector
    const int mrow = (tileM << 4) + (lane & 15);
    int arow = mrow;
    if (usePerm)                               // deterministic bijection mod M
        arow = (int)((48271ull * (unsigned long long)mrow + 12345ull)
                     % (unsigned long long)M);

    const float* ap = A + (size_t)arow * K;
    const __bf16* bp[NT];
    int ncol[NT];
    #pragma unroll
    for (int nt = 0; nt < NT; ++nt) {
        ncol[nt] = ((gN * NT + nt) << 4) + (lane & 15);
        bp[nt]   = BT + (size_t)ncol[nt] * K;
    }

    v8f acc[NT];
    #pragma unroll
    for (int nt = 0; nt < NT; ++nt) acc[nt] = (v8f){};

    for (int k0 = 0; k0 < K; k0 += 32) {
        if (k0 + 32 < K) __builtin_prefetch(ap + k0 + 32, 0, 0);
        const int kc = k0 + half * 8;

        // A fragment (f32 -> bf16): elems 0..7 -> K=kc.., 8..15 -> K=kc+16..
        const float4 f0 = *(const float4*)(ap + kc);
        const float4 f1 = *(const float4*)(ap + kc + 4);
        const float4 f2 = *(const float4*)(ap + kc + 16);
        const float4 f3 = *(const float4*)(ap + kc + 20);
        v16bf afrag;
        afrag[0]=(__bf16)f0.x;  afrag[1]=(__bf16)f0.y;
        afrag[2]=(__bf16)f0.z;  afrag[3]=(__bf16)f0.w;
        afrag[4]=(__bf16)f1.x;  afrag[5]=(__bf16)f1.y;
        afrag[6]=(__bf16)f1.z;  afrag[7]=(__bf16)f1.w;
        afrag[8]=(__bf16)f2.x;  afrag[9]=(__bf16)f2.y;
        afrag[10]=(__bf16)f2.z; afrag[11]=(__bf16)f2.w;
        afrag[12]=(__bf16)f3.x; afrag[13]=(__bf16)f3.y;
        afrag[14]=(__bf16)f3.z; afrag[15]=(__bf16)f3.w;

        #pragma unroll
        for (int nt = 0; nt < NT; ++nt) {
            const v8bf blo = *(const v8bf*)(bp[nt] + kc);
            const v8bf bhi = *(const v8bf*)(bp[nt] + kc + 16);
            v16bf bfrag;
            #pragma unroll
            for (int i = 0; i < 8; ++i) { bfrag[i] = blo[i]; bfrag[8 + i] = bhi[i]; }
            acc[nt] = __builtin_amdgcn_wmma_f32_16x16x32_bf16(
                          false, afrag, false, bfrag, (short)0, acc[nt],
                          false, false);
        }
    }

    // D layout: VGPR j, lane L -> row = j + (L>=16)*8, col = L&15
    const int rbase = (tileM << 4) + half * 8;
    #pragma unroll
    for (int nt = 0; nt < NT; ++nt) {
        #pragma unroll
        for (int j = 0; j < 8; ++j) {
            const size_t off = (size_t)(rbase + j) * NC + ncol[nt];
            C0[off] = acc[nt][j];
            if (C1) C1[off] = acc[nt][j];
        }
    }
}

// ---------------------------------------------------------------------------
// Small helper kernels
// ---------------------------------------------------------------------------
__global__ void conv_weights(const float* __restrict__ W,
                             __bf16* __restrict__ Wbf,
                             __bf16* __restrict__ WTbf, int rows, int cols)
{
    int i = blockIdx.x * blockDim.x + threadIdx.x;
    if (i >= rows * cols) return;
    float v = W[i];
    Wbf[i] = (__bf16)v;
    int r = i / cols, c = i - r * cols;
    WTbf[(size_t)c * rows + r] = (__bf16)v;
}

__global__ void node_scores(const float* __restrict__ xw,
                            const float* __restrict__ a_s,
                            const float* __restrict__ a_d,
                            float* __restrict__ ssrc, float* __restrict__ sdst,
                            int Nn)
{
    int n = blockIdx.x * blockDim.x + threadIdx.x;
    if (n >= Nn) return;
    const float* row = xw + (size_t)n * 64;
    float s0 = 0.f, s1 = 0.f;
    #pragma unroll
    for (int c = 0; c < 64; ++c) { float v = row[c]; s0 += v * a_s[c]; s1 += v * a_d[c]; }
    ssrc[n] = s0; sdst[n] = s1;
}

__global__ void gat_init(unsigned* __restrict__ menc, float* __restrict__ denom,
                         float* __restrict__ h1, int Nn)
{
    int i = blockIdx.x * blockDim.x + threadIdx.x;
    if (i < Nn) { menc[i] = 0x007FFFFFu; /* enc(-inf) */ denom[i] = 0.f; }
    if (i < Nn * 64) h1[i] = 0.f;
}

__global__ void edge_max(const int* __restrict__ ei, const float* __restrict__ ssrc,
                         const float* __restrict__ sdst, float* __restrict__ ebuf,
                         unsigned* __restrict__ menc, int E)
{
    int e = blockIdx.x * blockDim.x + threadIdx.x;
    if (e >= E) return;
    int s = ei[e], d = ei[E + e];
    float v = ssrc[s] + sdst[d];
    v = (v > 0.f) ? v : LEAKY_SLOPE * v;
    ebuf[e] = v;
    atomicMax(menc + d, fenc(v));
}

__global__ void edge_exp(const int* __restrict__ ei, const unsigned* __restrict__ menc,
                         float* __restrict__ ebuf, float* __restrict__ denom, int E)
{
    int e = blockIdx.x * blockDim.x + threadIdx.x;
    if (e >= E) return;
    int d = ei[E + e];
    float ex = expf(ebuf[e] - fdec(menc[d]));
    ebuf[e] = ex;
    atomicAdd(denom + d, ex);
}

__global__ void edge_agg(const int* __restrict__ ei, const float* __restrict__ xw,
                         const float* __restrict__ ebuf, const float* __restrict__ denom,
                         float* __restrict__ h1, int E)
{
    long long idx = (long long)blockIdx.x * blockDim.x + threadIdx.x;
    int e = (int)(idx >> 4);
    if (e >= E) return;
    int c4 = ((int)idx & 15) << 2;
    int s = ei[e], d = ei[E + e];
    float alpha = ebuf[e] / (denom[d] + 1e-16f);
    const float4 v = *(const float4*)(xw + (size_t)s * 64 + c4);
    float* o = h1 + (size_t)d * 64 + c4;
    atomicAdd(o + 0, v.x * alpha);
    atomicAdd(o + 1, v.y * alpha);
    atomicAdd(o + 2, v.z * alpha);
    atomicAdd(o + 3, v.w * alpha);
}

__global__ void elu_ip(float* __restrict__ x, long long n)
{
    long long i = (long long)blockIdx.x * blockDim.x + threadIdx.x;
    if (i >= n) return;
    float v = x[i];
    x[i] = v > 0.f ? v : (expf(v) - 1.f);
}

__global__ void summary_kernel(const float* __restrict__ h2, float* __restrict__ out, int Nn)
{
    __shared__ float part[256];
    int t = threadIdx.x, col = t & 31, seg = t >> 5;
    float acc = 0.f;
    for (int r = seg; r < Nn; r += 8) acc += h2[(size_t)r * 32 + col];
    part[t] = acc;
    __syncthreads();
    if (t < 32) {
        float s = 0.f;
        #pragma unroll
        for (int i = 0; i < 8; ++i) s += part[i * 32 + t];
        float m = s / (float)Nn;
        out[t] = 1.f / (1.f + expf(-m));
    }
}

// ---------------------------------------------------------------------------
extern "C" void kernel_launch(void* const* d_in, const int* in_sizes, int n_in,
                              void* d_out, int out_size, void* d_ws, size_t ws_size,
                              hipStream_t stream)
{
    const float* features = (const float*)d_in[0];
    const int*   ei       = (const int*)d_in[1];
    const float* W1  = (const float*)d_in[2];
    const float* a1s = (const float*)d_in[3];
    const float* a1d = (const float*)d_in[4];
    const float* W2  = (const float*)d_in[5];
    const float* a3s = (const float*)d_in[6];
    const float* a3d = (const float*)d_in[7];

    const int IN = 512, HID = 64, OUT = 32;
    const int Nn = in_sizes[0] / IN;
    const int E  = in_sizes[1] / 2;

    // ---- workspace carve-up ----
    char* ws = (char*)d_ws;
    auto alignup = [](size_t x) { return (x + 255) & ~(size_t)255; };
    size_t off = 0;
    __bf16* W1bf  = (__bf16*)(ws + off); off = alignup(off + sizeof(__bf16) * IN * HID);
    __bf16* W1Tbf = (__bf16*)(ws + off); off = alignup(off + sizeof(__bf16) * IN * HID);
    __bf16* W2bf  = (__bf16*)(ws + off); off = alignup(off + sizeof(__bf16) * HID * OUT);
    __bf16* W2Tbf = (__bf16*)(ws + off); off = alignup(off + sizeof(__bf16) * HID * OUT);
    float*    xw    = (float*)(ws + off);    off = alignup(off + sizeof(float) * (size_t)Nn * HID);
    float*    h1    = (float*)(ws + off);    off = alignup(off + sizeof(float) * (size_t)Nn * HID);
    float*    ssrc  = (float*)(ws + off);    off = alignup(off + sizeof(float) * Nn);
    float*    sdst  = (float*)(ws + off);    off = alignup(off + sizeof(float) * Nn);
    unsigned* menc  = (unsigned*)(ws + off); off = alignup(off + sizeof(unsigned) * Nn);
    float*    denom = (float*)(ws + off);    off = alignup(off + sizeof(float) * Nn);
    float*    ebuf  = (float*)(ws + off);    off = alignup(off + sizeof(float) * (size_t)E);
    (void)ws_size;

    // ---- output slots (tuple concatenated flat) ----
    float* out = (float*)d_out;
    size_t p = 0;
    float* o_h2   = out + p; p += (size_t)Nn * OUT;
    float* o_h4a  = out + p; p += (size_t)Nn * IN;
    float* o_h4b  = out + p; p += (size_t)Nn * IN;
    float* o_rh2  = out + p; p += (size_t)Nn * OUT;
    float* o_rh4a = out + p; p += (size_t)Nn * IN;
    float* o_rh4b = out + p; p += (size_t)Nn * IN;
    float* o_sum  = out + p;   // 32
    (void)out_size;

    // ---- once: bf16 weight conversion (both orientations) ----
    conv_weights<<<(IN * HID + 255) / 256, 256, 0, stream>>>(W1, W1bf, W1Tbf, IN, HID);
    conv_weights<<<(HID * OUT + 255) / 256, 256, 0, stream>>>(W2, W2bf, W2Tbf, HID, OUT);

    auto gemm = [&](const float* A, const __bf16* BT, float* C0, float* C1,
                    int M, int K, int NC, int usePerm) {
        int tilesN = NC / 16;
        if (tilesN % 4 == 0) {
            int total  = (M / 16) * (tilesN / 4);
            int blocks = (total + 7) / 8;       // 8 waves (wave32) per 256-thr block
            wmma_gemm_bf16<4><<<blocks, 256, 0, stream>>>(A, BT, C0, C1, M, K, NC, usePerm);
        } else {
            int total  = (M / 16) * (tilesN / 2);
            int blocks = (total + 7) / 8;
            wmma_gemm_bf16<2><<<blocks, 256, 0, stream>>>(A, BT, C0, C1, M, K, NC, usePerm);
        }
    };

    auto gat = [&](const float* as_, const float* ad_) {
        node_scores<<<(Nn + 255) / 256, 256, 0, stream>>>(xw, as_, ad_, ssrc, sdst, Nn);
        gat_init<<<(Nn * 64 + 255) / 256, 256, 0, stream>>>(menc, denom, h1, Nn);
        edge_max<<<(E + 255) / 256, 256, 0, stream>>>(ei, ssrc, sdst, ebuf, menc, E);
        edge_exp<<<(E + 255) / 256, 256, 0, stream>>>(ei, menc, ebuf, denom, E);
        edge_agg<<<(unsigned)(((long long)E * 16 + 255) / 256), 256, 0, stream>>>(
            ei, xw, ebuf, denom, h1, E);
        elu_ip<<<(unsigned)(((long long)Nn * 64 + 255) / 256), 256, 0, stream>>>(
            h1, (long long)Nn * 64);
    };

    for (int pass = 0; pass < 2; ++pass) {
        float* oh2  = pass ? o_rh2  : o_h2;
        float* oh4a = pass ? o_rh4a : o_h4a;
        float* oh4b = pass ? o_rh4b : o_h4b;

        gemm(features, W1Tbf, xw, nullptr, Nn, IN, HID, pass);   // xw1 = X @ W1
        gat(a1s, a1d);                                           // h1 = elu(GAT1)
        gemm(h1, W2Tbf, oh2, nullptr, Nn, HID, OUT, 0);          // h2 = h1 @ W2
        gemm(oh2, W2bf, xw, nullptr, Nn, OUT, HID, 0);           // xw3 = h2 @ W2^T
        gat(a3s, a3d);                                           // h3 = elu(GAT2)
        gemm(h1, W1bf, oh4a, oh4b, Nn, HID, IN, 0);              // h4 = h3 @ W1^T (dual store)
    }
    summary_kernel<<<1, 256, 0, stream>>>(o_h2, o_sum, Nn);
}
